// MoleculeGNN_Fourier_5686536700281
// MI455X (gfx1250) — matrix operand order repaired and checked by reading, not compiled, not run on hardware
//
#include <hip/hip_runtime.h>

// ---------------------------------------------------------------------------
// MoleculeGNN_Fourier on MI455X (gfx1250, wave32, WMMA f16 16x16x32)
//
// proj[e,f] = x_i^T A_f x_j  ->  [Ex1056] x [1056x32] GEMM with A generated
// on the fly: chunk c (<32): A_row(m) = x_i[m,c] * x_j[m,:]; chunk 32: x_j
// (carries the bf bias). msg = emb @ Wo as a second WMMA GEMM; node-update
// MLP and classifier are WMMA tiles of 16 nodes. All weights pre-converted
// to f16 in B-fragment-swizzled order.
// ---------------------------------------------------------------------------

typedef _Float16 half16 __attribute__((ext_vector_type(16)));
typedef float    f32x8  __attribute__((ext_vector_type(8)));

#define kN     10000
#define kD     32
#define kH     128
#define kF     32
#define kC     10
#define kL     3
#define kERAW  160000
#define kETOT  170000            // kERAW + kN self loops
#define kTILE_E (kETOT / 16)     // 10625 (exact)
#define kTILE_N (kN / 16)        // 625  (exact)

// ---- workspace layout (bytes) ----
__host__ __device__ constexpr size_t alignup(size_t x) { return (x + 255) & ~(size_t)255; }
constexpr size_t OFF_XCUR = 0;
constexpr size_t OFF_AGG  = OFF_XCUR + alignup((size_t)kN * 32 * 4);
constexpr size_t OFF_CNT  = OFF_AGG  + alignup((size_t)kN * 32 * 4);
constexpr size_t OFF_INV  = OFF_CNT  + alignup((size_t)kN * 4);
constexpr size_t OFF_WFE  = OFF_INV  + alignup((size_t)kN * 4);
constexpr size_t OFF_WOE  = OFF_WFE  + alignup(66 * 512 * 2);
constexpr size_t OFF_W1E  = OFF_WOE  + alignup(12 * 512 * 2);
constexpr size_t OFF_W2E  = OFF_W1E  + alignup(16 * 512 * 2);
constexpr size_t OFF_WC1E = OFF_W2E  + alignup( 8 * 512 * 2);
constexpr size_t OFF_WC2E = OFF_WC1E + alignup( 8 * 512 * 2);

// dynamic LDS for the edge kernel
constexpr unsigned SMEM_EDGE = 66*512*2   /* WfE swz  67584 */
                             + 12*512*2   /* WoE swz  12288 */
                             + 2*16*32*4  /* xi f32    4096 */
                             + 2*16*32*2  /* xj f16    2048 */
                             + 2*16*192*2 /* emb f16  12288 */
                             + 2*16*4;    /* dst       128  */   // = 98432 B

// ---------------------------------------------------------------------------
// helpers
// ---------------------------------------------------------------------------
static __device__ __forceinline__ void wave_fence() {
  // intra-wave LDS producer->consumer ordering (wave32 lockstep; DS in-order,
  // but we must stop the compiler reordering and wait DScnt).
  __builtin_amdgcn_wave_barrier();
  asm volatile("s_wait_dscnt 0" ::: "memory");
  __builtin_amdgcn_wave_barrier();
}

static __device__ __forceinline__ f32x8 zero8() {
  f32x8 z = {0.f,0.f,0.f,0.f,0.f,0.f,0.f,0.f};
  return z;
}

static __device__ __forceinline__ f32x8 wmma(half16 a, half16 b, f32x8 c) {
  return __builtin_amdgcn_wmma_f32_16x16x32_f16(false, a, false, b, (short)0, c, false, false);
}

// A fragment (16x32 f16): lane holds row m=lane&15; halves 0..7 = K b0..b0+7,
// halves 8..15 = K 16+b0.. ; b0 = (lane<16)?0:8.  `row` points at row start,
// chunk selects K-chunk of 32 within a row of Ktot halves.
static __device__ __forceinline__ half16 afrag(const _Float16* row, int chunk, int b0) {
  union { half16 h; uint4 u[2]; } u;
  u.u[0] = *(const uint4*)(row + chunk * 32 + b0);
  u.u[1] = *(const uint4*)(row + chunk * 32 + 16 + b0);
  return u.h;
}

// B fragment pre-swizzled: frag `fr`, lane -> 16 contiguous halves.
static __device__ __forceinline__ half16 bfrag(const _Float16* swz, int fr, int lane) {
  const uint4* p = (const uint4*)(swz + ((size_t)((fr << 5) + lane) << 4));
  union { half16 h; uint4 u[2]; } u;
  u.u[0] = p[0];
  u.u[1] = p[1];
  return u.h;
}

// ---------------------------------------------------------------------------
// init / counts
// ---------------------------------------------------------------------------
__global__ void init_kernel(const float* __restrict__ x, float* __restrict__ xcur,
                            int* __restrict__ cnt) {
  int i = blockIdx.x * 256 + threadIdx.x;
  if (i < kN * 32) xcur[i] = x[i];
  if (i < kN) cnt[i] = 1;               // self loop
}

__global__ void count_kernel(const int* __restrict__ ei, int* __restrict__ cnt) {
  int e = blockIdx.x * 256 + threadIdx.x;
  if (e < kERAW) atomicAdd(&cnt[ei[kERAW + e]], 1);
}

__global__ void recip_kernel(const int* __restrict__ cnt, float* __restrict__ invc) {
  int i = blockIdx.x * 256 + threadIdx.x;
  if (i < kN) invc[i] = 1.0f / (float)cnt[i];
}

__global__ void zero_kernel(float* __restrict__ p, int n) {
  int i = blockIdx.x * 256 + threadIdx.x;
  if (i < n) p[i] = 0.0f;
}

// ---------------------------------------------------------------------------
// weight prep: f32 -> f16, B-fragment-swizzled order
// ---------------------------------------------------------------------------
// WfE: K=1056 (33 chunks), N=32 (2 tiles). k<1024: W'[(d',d),f] = Wf[d', f*32+d];
// k>=1024: bias rows bf[f*32 + (k-1024)].
__global__ void prep_wf_kernel(const float* __restrict__ Wf, const float* __restrict__ bf,
                               _Float16* __restrict__ dst) {
  int id = blockIdx.x * 64 + threadIdx.x;
  if (id >= 66 * 32) return;
  int fr = id >> 5, lane = id & 31;
  int c = fr >> 1, t = fr & 1;
  int kb = c * 32 + ((lane < 16) ? 0 : 8);
  int n  = t * 16 + (lane & 15);
  _Float16* o = dst + ((size_t)id << 4);
#pragma unroll
  for (int i = 0; i < 16; ++i) {
    int k = kb + i + ((i >= 8) ? 8 : 0);
    float v = (k < 1024) ? Wf[(k >> 5) * 1024 + n * 32 + (k & 31)]
                         : bf[n * 32 + (k - 1024)];
    o[i] = (_Float16)v;
  }
}

// plain row-major K x Nsrc weight -> swizzled frags; pads n>=Nvalid with 0.
__global__ void prep_plain_kernel(const float* __restrict__ src, _Float16* __restrict__ dst,
                                  int Nt, int Nsrc, int Nvalid, int totFr) {
  int id = blockIdx.x * 64 + threadIdx.x;
  if (id >= totFr * 32) return;
  int fr = id >> 5, lane = id & 31;
  int c = fr / Nt, t = fr % Nt;
  int kb = c * 32 + ((lane < 16) ? 0 : 8);
  int n  = t * 16 + (lane & 15);
  _Float16* o = dst + ((size_t)id << 4);
#pragma unroll
  for (int i = 0; i < 16; ++i) {
    int k = kb + i + ((i >= 8) ? 8 : 0);
    o[i] = (_Float16)((n < Nvalid) ? src[k * Nsrc + n] : 0.0f);
  }
}

// ---------------------------------------------------------------------------
// edge kernel: fused outer-product GEMM -> sin/cos emb -> msg GEMM -> scatter
// 2 waves / block, each wave owns its own 16-edge tiles; WfE/WoE shared in LDS.
// ---------------------------------------------------------------------------
__global__ __launch_bounds__(64) void edge_kernel(
    const float* __restrict__ xcur, const int* __restrict__ ei,
    const _Float16* __restrict__ WfE, const _Float16* __restrict__ WoE,
    const float* __restrict__ bo, float* __restrict__ agg) {
  extern __shared__ __align__(16) char smem[];
  _Float16* sWf  = (_Float16*)smem;                 // 66*512 halves
  _Float16* sWo  = sWf + 66 * 512;                  // 12*512 halves
  float*    sXi  = (float*)(sWo + 12 * 512);        // 2 waves * 16*32 f32
  _Float16* sXj  = (_Float16*)(sXi + 2 * 16 * 32);  // 2 waves * 16*32 f16
  _Float16* sEmb = sXj + 2 * 16 * 32;               // 2 waves * 16*192 f16
  int*      sDst = (int*)(sEmb + 2 * 16 * 192);     // 2 waves * 16

  const int tid = threadIdx.x;
  const int wave = tid >> 5, lane = tid & 31;

  {  // stage swizzled B matrices once per block (L2 resident, trivial cost)
    const uint4* s0 = (const uint4*)WfE;  uint4* d0 = (uint4*)sWf;
    for (int i = tid; i < 4224; i += 64) d0[i] = s0[i];
    const uint4* s1 = (const uint4*)WoE;  uint4* d1 = (uint4*)sWo;
    for (int i = tid; i < 768; i += 64) d1[i] = s1[i];
  }
  __syncthreads();

  const int nlo = lane & 15;
  const int b0  = (lane < 16) ? 0 : 8;
  const float bo0 = bo[nlo], bo1 = bo[nlo + 16];

  float*    myXi  = sXi  + wave * 512;
  _Float16* myXj  = sXj  + wave * 512;
  _Float16* myEmb = sEmb + wave * (16 * 192);
  int*      myDst = sDst + wave * 16;

  const int m2 = lane >> 1, p2 = lane & 1;   // staging role: edge m2, half p2

  for (int tile = blockIdx.x * 2 + wave; tile < kTILE_E; tile += gridDim.x * 2) {
    // ---- stage x_i (=x[dst], f32) and x_j (=x[src], f16) for 16 edges ----
    int e = tile * 16 + m2;
    int s, d;
    if (e < kERAW) { s = ei[e]; d = ei[kERAW + e]; }
    else           { s = e - kERAW; d = s; }
    if (p2 == 0) myDst[m2] = d;
    const float4* xs4 = (const float4*)(xcur + (size_t)s * 32 + p2 * 16);
    const float4* xd4 = (const float4*)(xcur + (size_t)d * 32 + p2 * 16);
    float4* xiDst = (float4*)(myXi + m2 * 32 + p2 * 16);
#pragma unroll
    for (int q = 0; q < 4; ++q) {
      float4 vs = xs4[q];
      xiDst[q] = xd4[q];
      int base = m2 * 32 + p2 * 16 + q * 4;
      myXj[base + 0] = (_Float16)vs.x;  myXj[base + 1] = (_Float16)vs.y;
      myXj[base + 2] = (_Float16)vs.z;  myXj[base + 3] = (_Float16)vs.w;
    }
    wave_fence();

    // ---- proj GEMM: K=1056 as 33 chunks; A chunk c = x_i[:,c] (.) x_j ----
    half16 xjf = afrag(myXj + nlo * 32, 0, b0);
    const float* xirow = myXi + nlo * 32;
    f32x8 c0 = zero8(), c1 = zero8();
#pragma unroll 4
    for (int c = 0; c < 32; ++c) {
      half16 a = xjf * (_Float16)xirow[c];
      c0 = wmma(a, bfrag(sWf, 2 * c + 0, lane), c0);
      c1 = wmma(a, bfrag(sWf, 2 * c + 1, lane), c1);
    }
    c0 = wmma(xjf, bfrag(sWf, 64, lane), c0);     // bias rows (bf)
    c1 = wmma(xjf, bfrag(sWf, 65, lane), c1);

    // ---- emb = [sin p, cos p, sin 2p, cos 2p, sin 4p, cos 4p] -> LDS ----
#pragma unroll
    for (int r = 0; r < 8; ++r) {
      int mm = r + b0;
      _Float16* er = myEmb + mm * 192;
      {
        float p = c0[r];
        float s1 = __sinf(p), cc1 = __cosf(p);
        float s2 = 2.f * s1 * cc1, cg2 = 1.f - 2.f * s1 * s1;
        float s4 = 2.f * s2 * cg2, cg4 = 1.f - 2.f * s2 * s2;
        er[0 * 32 + nlo] = (_Float16)s1;  er[1 * 32 + nlo] = (_Float16)cc1;
        er[2 * 32 + nlo] = (_Float16)s2;  er[3 * 32 + nlo] = (_Float16)cg2;
        er[4 * 32 + nlo] = (_Float16)s4;  er[5 * 32 + nlo] = (_Float16)cg4;
      }
      {
        float p = c1[r];
        float s1 = __sinf(p), cc1 = __cosf(p);
        float s2 = 2.f * s1 * cc1, cg2 = 1.f - 2.f * s1 * s1;
        float s4 = 2.f * s2 * cg2, cg4 = 1.f - 2.f * s2 * s2;
        er[0 * 32 + 16 + nlo] = (_Float16)s1;  er[1 * 32 + 16 + nlo] = (_Float16)cc1;
        er[2 * 32 + 16 + nlo] = (_Float16)s2;  er[3 * 32 + 16 + nlo] = (_Float16)cg2;
        er[4 * 32 + 16 + nlo] = (_Float16)s4;  er[5 * 32 + 16 + nlo] = (_Float16)cg4;
      }
    }
    wave_fence();

    // ---- msg GEMM: [16x192] @ [192x32] ----
    f32x8 d0 = zero8(), d1 = zero8();
#pragma unroll
    for (int c = 0; c < 6; ++c) {
      half16 a = afrag(myEmb + nlo * 192, c, b0);
      d0 = wmma(a, bfrag(sWo, 2 * c + 0, lane), d0);
      d1 = wmma(a, bfrag(sWo, 2 * c + 1, lane), d1);
    }

    // ---- scatter-add msg (+bo) into agg[dst] ----
#pragma unroll
    for (int r = 0; r < 8; ++r) {
      int mm = r + b0;
      float* ap = agg + (size_t)myDst[mm] * 32;
      unsafeAtomicAdd(ap + nlo,      d0[r] + bo0);
      unsafeAtomicAdd(ap + nlo + 16, d1[r] + bo1);
    }
    wave_fence();
  }
}

// ---------------------------------------------------------------------------
// node update: h=[x | agg/cnt] -> relu(h@W1+b1)@W2+b2 -> relu -> x += out
// ---------------------------------------------------------------------------
__global__ __launch_bounds__(32) void node_kernel(
    float* __restrict__ xcur, const float* __restrict__ agg,
    const float* __restrict__ invc,
    const _Float16* __restrict__ W1E, const _Float16* __restrict__ W2E,
    const float* __restrict__ b1, const float* __restrict__ b2) {
  __shared__ __align__(16) _Float16 sH[16 * 64];
  __shared__ __align__(16) _Float16 sM[16 * 128];
  const int lane = threadIdx.x, tile = blockIdx.x;
  const int m2 = lane >> 1, p2 = lane & 1;
  const int node = tile * 16 + m2;
  {
    const float* src = p2 ? (agg + (size_t)node * 32) : (xcur + (size_t)node * 32);
    float scale = p2 ? invc[node] : 1.0f;
    _Float16* dr = sH + m2 * 64 + p2 * 32;
#pragma unroll
    for (int j = 0; j < 32; ++j) dr[j] = (_Float16)(src[j] * scale);
  }
  wave_fence();

  const int nlo = lane & 15;
  const int b0  = (lane < 16) ? 0 : 8;

  f32x8 acc[8];
#pragma unroll
  for (int t = 0; t < 8; ++t) acc[t] = zero8();
#pragma unroll
  for (int c = 0; c < 2; ++c) {
    half16 a = afrag(sH + nlo * 64, c, b0);
#pragma unroll
    for (int t = 0; t < 8; ++t) acc[t] = wmma(a, bfrag(W1E, c * 8 + t, lane), acc[t]);
  }
#pragma unroll
  for (int t = 0; t < 8; ++t) {
    float bb = b1[t * 16 + nlo];
#pragma unroll
    for (int r = 0; r < 8; ++r) {
      float v = acc[t][r] + bb;
      v = v > 0.f ? v : 0.f;
      sM[(r + b0) * 128 + t * 16 + nlo] = (_Float16)v;
    }
  }
  wave_fence();

  f32x8 o0 = zero8(), o1 = zero8();
#pragma unroll
  for (int c = 0; c < 4; ++c) {
    half16 a = afrag(sM + nlo * 128, c, b0);
    o0 = wmma(a, bfrag(W2E, 2 * c + 0, lane), o0);
    o1 = wmma(a, bfrag(W2E, 2 * c + 1, lane), o1);
  }
  const float bb0 = b2[nlo], bb1 = b2[nlo + 16];
#pragma unroll
  for (int r = 0; r < 8; ++r) {
    int nd = tile * 16 + r + b0;
    float* xr = xcur + (size_t)nd * 32;
    float v0 = o0[r] + bb0;  v0 = v0 > 0.f ? v0 : 0.f;
    float v1 = o1[r] + bb1;  v1 = v1 > 0.f ? v1 : 0.f;
    xr[nlo]      += v0;                       // residual
    xr[nlo + 16] += v1;
  }
}

// ---------------------------------------------------------------------------
// classifier: relu(x@Wc1+bc1)@Wc2+bc2  (N=10 cols, padded to 16 in Wc2E)
// ---------------------------------------------------------------------------
__global__ __launch_bounds__(32) void cls_kernel(
    const float* __restrict__ xcur,
    const _Float16* __restrict__ Wc1E, const _Float16* __restrict__ Wc2E,
    const float* __restrict__ bc1, const float* __restrict__ bc2,
    float* __restrict__ out) {
  __shared__ __align__(16) _Float16 sH[16 * 32];
  __shared__ __align__(16) _Float16 sM[16 * 128];
  const int lane = threadIdx.x, tile = blockIdx.x;
  const int m2 = lane >> 1, p2 = lane & 1;
  const int node = tile * 16 + m2;
  {
    const float* src = xcur + (size_t)node * 32 + p2 * 16;
    _Float16* dr = sH + m2 * 32 + p2 * 16;
#pragma unroll
    for (int j = 0; j < 16; ++j) dr[j] = (_Float16)src[j];
  }
  wave_fence();

  const int nlo = lane & 15;
  const int b0  = (lane < 16) ? 0 : 8;

  f32x8 acc[8];
#pragma unroll
  for (int t = 0; t < 8; ++t) acc[t] = zero8();
  {
    half16 a = afrag(sH + nlo * 32, 0, b0);
#pragma unroll
    for (int t = 0; t < 8; ++t) acc[t] = wmma(a, bfrag(Wc1E, t, lane), acc[t]);
  }
#pragma unroll
  for (int t = 0; t < 8; ++t) {
    float bb = bc1[t * 16 + nlo];
#pragma unroll
    for (int r = 0; r < 8; ++r) {
      float v = acc[t][r] + bb;
      v = v > 0.f ? v : 0.f;
      sM[(r + b0) * 128 + t * 16 + nlo] = (_Float16)v;
    }
  }
  wave_fence();

  f32x8 o0 = zero8();
#pragma unroll
  for (int c = 0; c < 4; ++c) {
    half16 a = afrag(sM + nlo * 128, c, b0);
    o0 = wmma(a, bfrag(Wc2E, c, lane), o0);
  }
  if (nlo < kC) {
    float bb = bc2[nlo];
#pragma unroll
    for (int r = 0; r < 8; ++r) {
      int nd = tile * 16 + r + b0;
      out[(size_t)nd * kC + nlo] = o0[r] + bb;
    }
  }
}

// ---------------------------------------------------------------------------
// host-side launch
// ---------------------------------------------------------------------------
extern "C" void kernel_launch(void* const* d_in, const int* in_sizes, int n_in,
                              void* d_out, int out_size, void* d_ws, size_t ws_size,
                              hipStream_t stream) {
  (void)in_sizes; (void)n_in; (void)out_size; (void)ws_size;
  const float* x   = (const float*)d_in[0];
  const int*   ei  = (const int*)d_in[1];
  const float* Wf  = (const float*)d_in[2];
  const float* bf  = (const float*)d_in[3];
  const float* Wo  = (const float*)d_in[4];
  const float* bo  = (const float*)d_in[5];
  const float* W1  = (const float*)d_in[6];
  const float* b1  = (const float*)d_in[7];
  const float* W2  = (const float*)d_in[8];
  const float* b2  = (const float*)d_in[9];
  const float* Wc1 = (const float*)d_in[10];
  const float* bc1 = (const float*)d_in[11];
  const float* Wc2 = (const float*)d_in[12];
  const float* bc2 = (const float*)d_in[13];

  char* ws = (char*)d_ws;
  float*     xcur = (float*)(ws + OFF_XCUR);
  float*     agg  = (float*)(ws + OFF_AGG);
  int*       cnt  = (int*)(ws + OFF_CNT);
  float*     invc = (float*)(ws + OFF_INV);
  _Float16*  WfE  = (_Float16*)(ws + OFF_WFE);
  _Float16*  WoE  = (_Float16*)(ws + OFF_WOE);
  _Float16*  W1E  = (_Float16*)(ws + OFF_W1E);
  _Float16*  W2E  = (_Float16*)(ws + OFF_W2E);
  _Float16*  Wc1E = (_Float16*)(ws + OFF_WC1E);
  _Float16*  Wc2E = (_Float16*)(ws + OFF_WC2E);

  init_kernel<<<(kN * 32 + 255) / 256, 256, 0, stream>>>(x, xcur, cnt);
  count_kernel<<<(kERAW + 255) / 256, 256, 0, stream>>>(ei, cnt);
  recip_kernel<<<(kN + 255) / 256, 256, 0, stream>>>(cnt, invc);

  // classifier weights (layer-independent)
  prep_plain_kernel<<<4, 64, 0, stream>>>(Wc1, Wc1E, 8, 128, 128, 8);
  prep_plain_kernel<<<2, 64, 0, stream>>>(Wc2, Wc2E, 1, 10, 10, 4);

  for (int l = 0; l < kL; ++l) {
    prep_wf_kernel<<<33, 64, 0, stream>>>(Wf + (size_t)l * 32 * 1024,
                                          bf + (size_t)l * 1024, WfE);
    prep_plain_kernel<<<6, 64, 0, stream>>>(Wo + (size_t)l * 192 * 32, WoE, 2, 32, 32, 12);
    prep_plain_kernel<<<8, 64, 0, stream>>>(W1 + (size_t)l * 64 * 128, W1E, 8, 128, 128, 16);
    prep_plain_kernel<<<4, 64, 0, stream>>>(W2 + (size_t)l * 128 * 32, W2E, 2, 32, 32, 8);

    zero_kernel<<<(kN * 32 + 255) / 256, 256, 0, stream>>>(agg, kN * 32);
    edge_kernel<<<160, 64, SMEM_EDGE, stream>>>(xcur, ei, WfE, WoE,
                                                bo + (size_t)l * 32, agg);
    node_kernel<<<kTILE_N, 32, 0, stream>>>(xcur, agg, invc, W1E, W2E,
                                            b1 + (size_t)l * 128, b2 + (size_t)l * 32);
  }

  cls_kernel<<<kTILE_N, 32, 0, stream>>>(xcur, Wc1E, Wc2E, bc1, bc2, (float*)d_out);
}
